// FreqFusion1x_41162966565204
// MI455X (gfx1250) — matrix operand order repaired and checked by reading, not compiled, not run on hardware
//
#include <hip/hip_runtime.h>
#include <hip/hip_bf16.h>

typedef __attribute__((ext_vector_type(16))) _Float16 v16h;
typedef __attribute__((ext_vector_type(8)))  _Float16 v8h;
typedef __attribute__((ext_vector_type(2)))  _Float16 h2;
typedef __attribute__((ext_vector_type(8)))  float    v8f;

#define HH   128
#define WWID 128
#define HWSZ (HH * WWID)
#define NB   4
#define CINF 256
#define CCC  64

static __device__ __forceinline__ v8f wmma_f16(const v16h& a, const v16h& b, const v8f& c) {
    // D = A(16x32 f16) * B(32x16 f16) + C(16x16 f32)
    return __builtin_amdgcn_wmma_f32_16x16x32_f16(false, a, false, b, (short)0, c, false, false);
}

static __device__ __forceinline__ v16h cat8(v8h lo, v8h hi) {
    return __builtin_shufflevector(lo, hi, 0, 1, 2, 3, 4, 5, 6, 7,
                                           8, 9, 10, 11, 12, 13, 14, 15);
}

// ---------------------------------------------------------------------------
// Tensor Data Mover: 2D tile load Global -> LDS (D# per CDNA5 ISA).
// Group0: count=1 | lds_addr | global_addr[56:0] | type=2.
// Group1: data_size | tensor_dim0/1 | tile_dim0/1 | tensor_dim0_stride.
// Issued wave-uniform; completion via s_wait_tensorcnt.
// ---------------------------------------------------------------------------
#if __has_builtin(__builtin_amdgcn_tensor_load_to_lds) && __has_builtin(__builtin_amdgcn_s_wait_tensorcnt)
#define HAVE_TDM 1
typedef unsigned int u32x4 __attribute__((ext_vector_type(4)));
typedef int          i32x4 __attribute__((ext_vector_type(4)));
typedef int          i32x8 __attribute__((ext_vector_type(8)));

static __device__ __forceinline__ void tdm_load_2d(
    const void* gsrc, void* lds_dst, unsigned ds_code /*0=1B,1=2B,2=4B*/,
    unsigned tile_d0, unsigned tile_d1,
    unsigned long long stride0_elems, unsigned tdim0, unsigned tdim1)
{
    unsigned long long ga = (unsigned long long)gsrc;
    unsigned lds = (unsigned)(unsigned long long)lds_dst;  // low 32 bits = LDS offset
    u32x4 g0;
    g0[0] = 1u;                                            // count=1 (valid user D#)
    g0[1] = lds;                                           // lds_addr
    g0[2] = (unsigned)(ga & 0xffffffffu);                  // global_addr[31:0]
    g0[3] = (unsigned)((ga >> 32) & 0x01ffffffu) | (2u << 30);  // [56:32] | type=2
    i32x8 g1;
    g1[0] = (int)(ds_code << 16);                          // wg_mask=0 | data_size
    g1[1] = (int)((tdim0 & 0xffffu) << 16);                // abar=0 | tensor_dim0 lo
    g1[2] = (int)((tdim0 >> 16) | ((tdim1 & 0xffffu) << 16));
    g1[3] = (int)((tdim1 >> 16) | ((tile_d0 & 0xffffu) << 16));
    g1[4] = (int)(tile_d1 & 0xffffu);                      // tile_dim1 | tile_dim2=0
    g1[5] = (int)(unsigned)(stride0_elems & 0xffffffffu);  // dim0 stride lo
    g1[6] = (int)(unsigned)((stride0_elems >> 32) & 0xffffu);
    g1[7] = 0;
    i32x4 z4 = {0, 0, 0, 0};
#if __clang_major__ >= 23
    i32x8 z8 = {0, 0, 0, 0, 0, 0, 0, 0};
    __builtin_amdgcn_tensor_load_to_lds(g0, g1, z4, z4, z8, 0);
#else
    __builtin_amdgcn_tensor_load_to_lds(g0, g1, z4, z4, 0);
#endif
}
#else
#define HAVE_TDM 0
#endif

// -------------------------------------------------------------------------
// Kernel 1: fused 1x1 compress convs (hr and lr), WMMA GEMM [64x256]*[256xP]
// writes cf16 = concat(chr, clr) in f16 (NCHW, 128 ch) and chr in f32.
// Activation tile staged transposed [px][c] so B fragments are contiguous
// (2x ds_load_b128 per fragment instead of 16x ds_load_u16).
// -------------------------------------------------------------------------
__global__ __launch_bounds__(256) void k_compress(
    const float* __restrict__ hr, const float* __restrict__ lr,
    const float* __restrict__ hrw, const float* __restrict__ hrb,
    const float* __restrict__ lrw, const float* __restrict__ lrb,
    float* __restrict__ chr32, _Float16* __restrict__ cf16)
{
    const int is_lr = blockIdx.y;
    const float* src  = is_lr ? lr  : hr;
    const float* wgt  = is_lr ? lrw : hrw;
    const float* bias = is_lr ? lrb : hrb;

    const int tid  = threadIdx.x;
    const int lane = tid & 31;
    const int wave = tid >> 5;
    const int pixBase = blockIdx.x * 128;   // 8 waves x 16 pixels
    const int n  = pixBase / HWSZ;          // 128 | HWSZ -> whole block same image
    const int hw = pixBase % HWSZ;

    __shared__ _Float16 sW[CCC * 32];       // [64 oc][32 c]
    __shared__ _Float16 sX[128 * 32];       // transposed: [128 px][32 c]

    v8f acc[4] = {};                        // 4 M-tiles cover oc 0..63

    for (int k0 = 0; k0 < CINF; k0 += 32) {
        __syncthreads();
        // weights: pack channel pairs -> one aligned 32-bit LDS store
        for (int i = tid; i < CCC * 16; i += 256) {
            int oc = i >> 4, cp = (i & 15) * 2;
            h2 v;
            v[0] = (_Float16)wgt[(size_t)oc * CINF + k0 + cp];
            v[1] = (_Float16)wgt[(size_t)oc * CINF + k0 + cp + 1];
            *(h2*)&sW[oc * 32 + cp] = v;
        }
        // activations: two channels per thread (coalesced along px per channel)
        for (int i = tid; i < 16 * 128; i += 256) {
            int px = i & 127, cp = (i >> 7) * 2;
            const float* p = &src[((size_t)n * CINF + k0 + cp) * HWSZ + hw + px];
            h2 v;
            v[0] = (_Float16)p[0];
            v[1] = (_Float16)p[HWSZ];
            if (k0 + 32 < CINF)
                __builtin_prefetch(p + (size_t)32 * HWSZ, 0, 0);  // global_prefetch_b8
            *(h2*)&sX[px * 32 + cp] = v;
        }
        __syncthreads();

        // B fragment: 32(K)x16(N); two contiguous 8-half chunks (K and K+16)
        const int col = wave * 16 + (lane & 15);
        const _Float16* pb = &sX[col * 32 + ((lane >> 4) << 3)];
        v16h b = cat8(*(const v8h*)pb, *(const v8h*)(pb + 16));

        #pragma unroll
        for (int t = 0; t < 4; ++t) {
            // A fragment: 16(M)x32(K); two contiguous 8-half chunks (K and K+16)
            const _Float16* pa = &sW[(t * 16 + (lane & 15)) * 32 + ((lane >> 4) << 3)];
            v16h a = cat8(*(const v8h*)pa, *(const v8h*)(pa + 16));
            acc[t] = wmma_f16(a, b, acc[t]);
        }
    }

    const int px = wave * 16 + (lane & 15);
    #pragma unroll
    for (int t = 0; t < 4; ++t) {
        #pragma unroll
        for (int v = 0; v < 8; ++v) {
            int oc = t * 16 + ((lane >> 4) << 3) + v;   // D layout: M=(laneHi)*8+v
            float val = acc[t][v] + bias[oc];
            if (!is_lr)
                chr32[((size_t)n * CCC + oc) * HWSZ + hw + px] = val;
            cf16[((size_t)n * 128 + (is_lr ? CCC : 0) + oc) * HWSZ + hw + px] = (_Float16)val;
        }
    }
}

// -------------------------------------------------------------------------
// Kernel 2: 3x3 mask conv as 9 shifted 1x1 WMMA GEMMs. Block = one image row
// (128 px = 8 waves x 16). Activation tile transposed [col 130][c 32].
// NT = number of 16-wide oc tiles (1 for Cout=9, 2 for Cout=25).
// -------------------------------------------------------------------------
__global__ __launch_bounds__(256) void k_mask_conv(
    const _Float16* __restrict__ src, const float* __restrict__ wgt,
    const float* __restrict__ bias, int Cout, int NT, float* __restrict__ outraw)
{
    const int tid  = threadIdx.x;
    const int lane = tid & 31;
    const int wave = tid >> 5;
    const int row  = blockIdx.x;
    const int n = row >> 7, h = row & (HH - 1);

    __shared__ _Float16 sW[9 * 32 * 32];    // [tap][oc 32][c 32]  (18 KB)
    __shared__ _Float16 sX[130 * 32];       // transposed: [w halo 130][c 32]

    v8f acc0 = {}, acc1 = {};

    for (int c0 = 0; c0 < 128; c0 += 32) {
        for (int dy = 0; dy < 3; ++dy) {
            __syncthreads();
            if (dy == 0) {
                for (int i = tid; i < 9 * 32 * 16; i += 256) {
                    int tap = i >> 9, rem = i & 511;
                    int oc = rem >> 4, cp = (rem & 15) * 2;
                    h2 v = {(_Float16)0.f, (_Float16)0.f};
                    if (oc < Cout) {
                        v[0] = (_Float16)wgt[((size_t)oc * 128 + c0 + cp) * 9 + tap];
                        v[1] = (_Float16)wgt[((size_t)oc * 128 + c0 + cp + 1) * 9 + tap];
                    }
                    *(h2*)&sW[tap * 1024 + oc * 32 + cp] = v;
                }
            }
            const int hh = h + dy - 1;
            for (int i = tid; i < 16 * 130; i += 256) {
                int col = i % 130, cp = (i / 130) * 2;
                int wp = col - 1;
                h2 v = {(_Float16)0.f, (_Float16)0.f};
                if (hh >= 0 && hh < HH && wp >= 0 && wp < WWID) {
                    const _Float16* p = &src[((size_t)n * 128 + c0 + cp) * HWSZ + hh * WWID + wp];
                    v[0] = p[0];
                    v[1] = p[HWSZ];
                }
                *(h2*)&sX[col * 32 + cp] = v;
            }
            __syncthreads();

            for (int dx = 0; dx < 3; ++dx) {
                const int tap = dy * 3 + dx;
                const int col = wave * 16 + (lane & 15) + dx;
                const _Float16* pb = &sX[col * 32 + ((lane >> 4) << 3)];
                v16h b = cat8(*(const v8h*)pb, *(const v8h*)(pb + 16));

                const _Float16* pa0 = &sW[tap * 1024 + (lane & 15) * 32 + ((lane >> 4) << 3)];
                v16h a0 = cat8(*(const v8h*)pa0, *(const v8h*)(pa0 + 16));
                acc0 = wmma_f16(a0, b, acc0);
                if (NT > 1) {   // uniform branch: EXEC stays all-ones for WMMA
                    const _Float16* pa1 = pa0 + 16 * 32;
                    v16h a1 = cat8(*(const v8h*)pa1, *(const v8h*)(pa1 + 16));
                    acc1 = wmma_f16(a1, b, acc1);
                }
            }
        }
    }

    const int px = wave * 16 + (lane & 15);
    #pragma unroll
    for (int v = 0; v < 8; ++v) {
        int oc = ((lane >> 4) << 3) + v;
        if (oc < Cout)
            outraw[((size_t)n * Cout + oc) * HWSZ + h * WWID + px] = acc0[v] + bias[oc];
        int oc1 = 16 + oc;
        if (NT > 1 && oc1 < Cout)
            outraw[((size_t)n * Cout + oc1) * HWSZ + h * WWID + px] = acc1[v] + bias[oc1];
    }
}

// -------------------------------------------------------------------------
// Kernel 3: softmax over k*k taps, modulate by hamming, renormalize.
// Fused: w_i = exp(v_i - max) * ham_i / sum_j(exp(v_j - max) * ham_j)
// -------------------------------------------------------------------------
template <int K2>
__global__ __launch_bounds__(256) void k_normalize(
    const float* __restrict__ raw, const float* __restrict__ ham,
    _Float16* __restrict__ o16, float* __restrict__ o32)
{
    const int p = blockIdx.x * blockDim.x + threadIdx.x;  // over NB*HWSZ
    const int n  = p >> 14;
    const int hw = p & (HWSZ - 1);

    float v[K2];
    float mx = -1e30f;
    #pragma unroll
    for (int i = 0; i < K2; ++i) {
        v[i] = raw[((size_t)n * K2 + i) * HWSZ + hw];
        mx = fmaxf(mx, v[i]);
    }
    float s = 0.f;
    #pragma unroll
    for (int i = 0; i < K2; ++i) {
        v[i] = __expf(v[i] - mx) * ham[i];
        s += v[i];
    }
    const float inv = 1.f / s;
    #pragma unroll
    for (int i = 0; i < K2; ++i) {
        float r = v[i] * inv;
        o16[((size_t)n * K2 + i) * HWSZ + hw] = (_Float16)r;
        if (o32) o32[((size_t)n * K2 + i) * HWSZ + hw] = r;
    }
}

// -------------------------------------------------------------------------
// Kernel 4: stage-1 CARAFE. chr_s1 = 2*chr - carafe3(chr16, mH1),
// clr_s1 = carafe5(clr16, mL1); both written f16 into cf1 (fused concat).
// Mask row tiles ([k*k][128], dim0 stride = HW) loaded via the TDM.
// -------------------------------------------------------------------------
__global__ __launch_bounds__(128) void k_carafe_s1(
    const _Float16* __restrict__ cf16, const float* __restrict__ chr32,
    const _Float16* __restrict__ mH16, const _Float16* __restrict__ mL16,
    _Float16* __restrict__ cf1)
{
    const int row = blockIdx.x;
    const int n = row >> 7, h = row & (HH - 1);
    const int w = threadIdx.x;

    __shared__ _Float16 smH[9 * 128];
    __shared__ _Float16 smL[25 * 128];
    __shared__ _Float16 sx3[3 * 130];
    __shared__ _Float16 sx5[5 * 132];

#if HAVE_TDM
    if (w < 32) {   // one wave issues; others sync on the barrier below
        tdm_load_2d(&mH16[(size_t)n * 9 * HWSZ + h * WWID], smH, 1,
                    128, 9, HWSZ, HWSZ, 9);
        tdm_load_2d(&mL16[(size_t)n * 25 * HWSZ + h * WWID], smL, 1,
                    128, 25, HWSZ, HWSZ, 25);
        __builtin_amdgcn_s_wait_tensorcnt(0);
    }
#else
    for (int i = w; i < 9 * 128; i += 128) {
        int t = i >> 7, c = i & 127;
        smH[i] = mH16[((size_t)n * 9 + t) * HWSZ + h * WWID + c];
    }
    for (int i = w; i < 25 * 128; i += 128) {
        int t = i / 128, c = i % 128;
        smL[i] = mL16[((size_t)n * 25 + t) * HWSZ + h * WWID + c];
    }
#endif

    // highpass residual on compressed hr (channels 0..63 of cf16)
    for (int c = 0; c < CCC; ++c) {
        __syncthreads();
        for (int i = w; i < 3 * 130; i += 128) {
            int r = i / 130, col = i % 130;
            int hh = h + r - 1, wp = col - 1;
            sx3[i] = (hh >= 0 && hh < HH && wp >= 0 && wp < WWID)
                   ? cf16[((size_t)n * 128 + c) * HWSZ + hh * WWID + wp] : (_Float16)0.f;
        }
        __syncthreads();
        float acc = 0.f;
        #pragma unroll
        for (int dy = 0; dy < 3; ++dy)
            #pragma unroll
            for (int dx = 0; dx < 3; ++dx)
                acc += (float)sx3[dy * 130 + w + dx] * (float)smH[(dy * 3 + dx) * 128 + w];
        float val = 2.f * chr32[((size_t)n * CCC + c) * HWSZ + h * WWID + w] - acc;
        cf1[((size_t)n * 128 + c) * HWSZ + h * WWID + w] = (_Float16)val;
    }

    // lowpass smoothing of compressed lr (channels 64..127 of cf16)
    for (int c = 0; c < CCC; ++c) {
        __syncthreads();
        for (int i = w; i < 5 * 132; i += 128) {
            int r = i / 132, col = i % 132;
            int hh = h + r - 2, wp = col - 2;
            sx5[i] = (hh >= 0 && hh < HH && wp >= 0 && wp < WWID)
                   ? cf16[((size_t)n * 128 + CCC + c) * HWSZ + hh * WWID + wp] : (_Float16)0.f;
        }
        __syncthreads();
        float acc = 0.f;
        #pragma unroll
        for (int dy = 0; dy < 5; ++dy)
            #pragma unroll
            for (int dx = 0; dx < 5; ++dx)
                acc += (float)sx5[dy * 132 + w + dx] * (float)smL[(dy * 5 + dx) * 128 + w];
        cf1[((size_t)n * 128 + CCC + c) * HWSZ + h * WWID + w] = (_Float16)acc;
    }
}

// -------------------------------------------------------------------------
// Kernel 5: final fused output. out = 2*hr - carafe3(hr f32, mH2 f32)
//                                    + carafe5(lr f16, mL2 f16)
// -------------------------------------------------------------------------
__global__ __launch_bounds__(128) void k_carafe_final(
    const float* __restrict__ hr, const float* __restrict__ lr,
    const float* __restrict__ mH32, const _Float16* __restrict__ mL16,
    float* __restrict__ out)
{
    const int row = blockIdx.x;
    const int n = row >> 7, h = row & (HH - 1);
    const int w = threadIdx.x;

    __shared__ float    smH[9 * 128];
    __shared__ _Float16 smL[25 * 128];
    __shared__ float    sh3[3 * 130];
    __shared__ _Float16 sl5[5 * 132];

#if HAVE_TDM
    if (w < 32) {
        tdm_load_2d(&mH32[(size_t)n * 9 * HWSZ + h * WWID], smH, 2,
                    128, 9, HWSZ, HWSZ, 9);
        tdm_load_2d(&mL16[(size_t)n * 25 * HWSZ + h * WWID], smL, 1,
                    128, 25, HWSZ, HWSZ, 25);
        __builtin_amdgcn_s_wait_tensorcnt(0);
    }
#else
    for (int i = w; i < 9 * 128; i += 128) {
        int t = i >> 7, c = i & 127;
        smH[i] = mH32[((size_t)n * 9 + t) * HWSZ + h * WWID + c];
    }
    for (int i = w; i < 25 * 128; i += 128) {
        int t = i / 128, c = i % 128;
        smL[i] = mL16[((size_t)n * 25 + t) * HWSZ + h * WWID + c];
    }
#endif

    for (int c = 0; c < CINF; ++c) {
        __syncthreads();
        for (int i = w; i < 3 * 130; i += 128) {
            int r = i / 130, col = i % 130;
            int hh = h + r - 1, wp = col - 1;
            sh3[i] = (hh >= 0 && hh < HH && wp >= 0 && wp < WWID)
                   ? hr[((size_t)n * CINF + c) * HWSZ + hh * WWID + wp] : 0.f;
        }
        for (int i = w; i < 5 * 132; i += 128) {
            int r = i / 132, col = i % 132;
            int hh = h + r - 2, wp = col - 2;
            sl5[i] = (hh >= 0 && hh < HH && wp >= 0 && wp < WWID)
                   ? (_Float16)lr[((size_t)n * CINF + c) * HWSZ + hh * WWID + wp] : (_Float16)0.f;
        }
        __syncthreads();
        float accH = 0.f, accL = 0.f;
        #pragma unroll
        for (int dy = 0; dy < 3; ++dy)
            #pragma unroll
            for (int dx = 0; dx < 3; ++dx)
                accH += sh3[dy * 130 + w + dx] * smH[(dy * 3 + dx) * 128 + w];
        #pragma unroll
        for (int dy = 0; dy < 5; ++dy)
            #pragma unroll
            for (int dx = 0; dx < 5; ++dx)
                accL += (float)sl5[dy * 132 + w + dx] * (float)smL[(dy * 5 + dx) * 128 + w];
        float hc = hr[((size_t)n * CINF + c) * HWSZ + h * WWID + w];
        out[((size_t)n * CINF + c) * HWSZ + h * WWID + w] = 2.f * hc - accH + accL;
    }
}

// -------------------------------------------------------------------------
extern "C" void kernel_launch(void* const* d_in, const int* in_sizes, int n_in,
                              void* d_out, int out_size, void* d_ws, size_t ws_size,
                              hipStream_t stream)
{
    (void)in_sizes; (void)n_in; (void)out_size; (void)ws_size;
    const float* hr   = (const float*)d_in[0];
    const float* lrf  = (const float*)d_in[1];
    const float* hrw  = (const float*)d_in[2];
    const float* hrb  = (const float*)d_in[3];
    const float* lrw  = (const float*)d_in[4];
    const float* lrb  = (const float*)d_in[5];
    const float* eL1w = (const float*)d_in[6];
    const float* eL1b = (const float*)d_in[7];
    const float* eL2w = (const float*)d_in[8];
    const float* eL2b = (const float*)d_in[9];
    const float* eH1w = (const float*)d_in[10];
    const float* eH1b = (const float*)d_in[11];
    const float* eH2w = (const float*)d_in[12];
    const float* eH2b = (const float*)d_in[13];
    const float* hamL = (const float*)d_in[14];
    const float* hamH = (const float*)d_in[15];
    float* out = (float*)d_out;

    char* ws = (char*)d_ws;
    size_t off = 0;
    auto alloc = [&](size_t bytes) -> void* {
        void* p = ws + off;
        off += (bytes + 255) & ~(size_t)255;
        return p;
    };
    _Float16* cf16  = (_Float16*)alloc(sizeof(_Float16) * (size_t)NB * 128 * HWSZ);
    _Float16* cf1   = (_Float16*)alloc(sizeof(_Float16) * (size_t)NB * 128 * HWSZ);
    float*    chr32 = (float*)   alloc(sizeof(float)    * (size_t)NB * CCC * HWSZ);
    float*    mrawH = (float*)   alloc(sizeof(float)    * (size_t)NB * 9  * HWSZ);
    float*    mrawL = (float*)   alloc(sizeof(float)    * (size_t)NB * 25 * HWSZ);
    _Float16* mH16  = (_Float16*)alloc(sizeof(_Float16) * (size_t)NB * 9  * HWSZ);
    _Float16* mL16  = (_Float16*)alloc(sizeof(_Float16) * (size_t)NB * 25 * HWSZ);
    float*    mH32  = (float*)   alloc(sizeof(float)    * (size_t)NB * 9  * HWSZ);

    const int rows = NB * HH;                 // 512
    const int ptot = (NB * HWSZ) / 256;       // 256 blocks for normalize

    // stage 0: compress + fused concat
    k_compress<<<dim3(512, 2), 256, 0, stream>>>(hr, lrf, hrw, hrb, lrw, lrb, chr32, cf16);

    // stage 1: masks from cf, carafe on compressed features
    k_mask_conv<<<rows, 256, 0, stream>>>(cf16, eH1w, eH1b, 9, 1, mrawH);
    k_normalize<9><<<ptot, 256, 0, stream>>>(mrawH, hamH, mH16, nullptr);
    k_mask_conv<<<rows, 256, 0, stream>>>(cf16, eL1w, eL1b, 25, 2, mrawL);
    k_normalize<25><<<ptot, 256, 0, stream>>>(mrawL, hamL, mL16, nullptr);
    k_carafe_s1<<<rows, 128, 0, stream>>>(cf16, chr32, mH16, mL16, cf1);

    // stage 2: masks from refined cf1, carafe on full-width features
    k_mask_conv<<<rows, 256, 0, stream>>>(cf1, eH2w, eH2b, 9, 1, mrawH);
    k_normalize<9><<<ptot, 256, 0, stream>>>(mrawH, hamH, mH16, mH32);
    k_mask_conv<<<rows, 256, 0, stream>>>(cf1, eL2w, eL2b, 25, 2, mrawL);
    k_normalize<25><<<ptot, 256, 0, stream>>>(mrawL, hamL, mL16, nullptr);
    k_carafe_final<<<rows, 128, 0, stream>>>(hr, lrf, mH32, mL16, out);
}
